// GeodesicAttention_36764920053927
// MI455X (gfx1250) — matrix operand order repaired and checked by reading, not compiled, not run on hardware
//
#include <hip/hip_runtime.h>
#include <hip/hip_bf16.h>

// ---------------------------------------------------------------------------
// Geodesic attention for MI455X (gfx1250, wave32, WMMA 16x16x32 f16/f32-acc)
//
//   Q = x Wq^T + bq ; K = x Wk^T + bk ; V = x Wv^T + bv        (f16 operands)
//   dist^2[q,k] = |Q_q|^2 + |K_k|^2 - 2 Q_q.K_k                (WMMA GEMM)
//   attn = softmax(-dist/16); out = (attn V) Wo^T + bo
//
// 5 launches total (launch latency dominates a ~3.2 GFLOP problem):
//   cvt_all -> proj_qkv (fused 3 GEMMs) -> norms (Q+K) -> attn -> proj_out
// ---------------------------------------------------------------------------

typedef __attribute__((ext_vector_type(16))) _Float16 v16h;
typedef __attribute__((ext_vector_type(8)))  _Float16 v8h;
typedef __attribute__((ext_vector_type(8)))  float    v8f;

#define BATCH 2
#define SEQ   1024
#define DIM   256
#define MROWS (BATCH * SEQ)   // 2048
#define NX    (MROWS * DIM)   // 524288 x elements
#define NW    (DIM * DIM)     // 65536 per weight

static __device__ __forceinline__ v16h cat8(v8h lo, v8h hi) {
  return __builtin_shufflevector(lo, hi, 0,1,2,3,4,5,6,7,8,9,10,11,12,13,14,15);
}
static __device__ __forceinline__ v8f wmma16(v16h a, v16h b, v8f c) {
  // D = A(16x32 f16) * B(32x16 f16) + C(16x16 f32)
  return __builtin_amdgcn_wmma_f32_16x16x32_f16(false, a, false, b,
                                                (short)0, c, false, false);
}

// ---------------- fused fp32 -> f16 conversion (x + 4 weights) -------------
__global__ __launch_bounds__(256) void cvt_all(
    const float* __restrict__ x,  const float* __restrict__ Wq,
    const float* __restrict__ Wk, const float* __restrict__ Wv,
    const float* __restrict__ Wo, _Float16* __restrict__ xh,
    _Float16* __restrict__ Wqh, _Float16* __restrict__ Wkh,
    _Float16* __restrict__ Wvh, _Float16* __restrict__ Woh) {
  int i = blockIdx.x * blockDim.x + threadIdx.x;
  if (i < NX) {
    xh[i] = (_Float16)x[i];
  } else {
    int j = i - NX;
    int sel = j >> 16, off = j & (NW - 1);
    const float* src = sel == 0 ? Wq : sel == 1 ? Wk : sel == 2 ? Wv : Wo;
    _Float16*    dst = sel == 0 ? Wqh : sel == 1 ? Wkh : sel == 2 ? Wvh : Woh;
    dst[off] = (_Float16)src[off];
  }
}

// --------------- row squared norms of Q and K in one launch ----------------
__global__ __launch_bounds__(256) void norms_kernel(
    const _Float16* __restrict__ Qh, const _Float16* __restrict__ Kh,
    float* __restrict__ qn, float* __restrict__ kn) {
  int row = (int)(blockIdx.x * 8 + (threadIdx.x >> 5));  // 0..4095
  int lane = threadIdx.x & 31;
  const _Float16* X = (row < MROWS ? Qh : Kh);
  float* out = (row < MROWS ? qn : kn);
  int r = row & (MROWS - 1);
  const _Float16* p = X + (size_t)r * DIM;
  float s = 0.f;
#pragma unroll
  for (int i = 0; i < DIM / 32; i++) {
    float v = (float)p[lane + i * 32];
    s += v * v;
  }
#pragma unroll
  for (int off = 16; off; off >>= 1) s += __shfl_xor(s, off, 32);
  if (lane == 0) out[r] = s;
}

// ---------------- one 16x16 WMMA tile of out = A * W^T + bias --------------
// A, W f16 row-major; B[k][n] = W[n][k] is K-contiguous.
// mode 0: outH row-major f16 ; mode 1: outH D-major f16 (V transposed)
// mode 2: outF row-major f32 (final projection)
static __device__ __forceinline__ void proj_tile(
    const _Float16* __restrict__ A, const _Float16* __restrict__ W,
    const float* __restrict__ bias, _Float16* __restrict__ outH,
    float* __restrict__ outF, int mode, int blk) {
  int wave = threadIdx.x >> 5, lane = threadIdx.x & 31;
  int wt = blk * 8 + wave;                 // 2048 wave-tiles
  int mt = wt >> 4, nt = wt & 15;          // 128 x 16 tiles of 16x16
  int row0 = mt * 16, n0 = nt * 16;
  int hs = lane >> 4, ln = lane & 15;

  const _Float16* Arow = A + (size_t)(row0 + ln) * DIM + hs * 16;
  const _Float16* Brow = W + (size_t)(n0 + ln) * DIM + hs * 16;

  v8f c = {};
#pragma unroll
  for (int k0 = 0; k0 < DIM; k0 += 32) {
    v8h alo = *(const v8h*)(Arow + k0);        // K = k0 + hs*16 .. +7
    v8h ahi = *(const v8h*)(Arow + k0 + 8);    // K = k0 + hs*16 + 8 ..
    v16h b  = *(const v16h*)(Brow + k0);       // 16 contiguous K at col n
    c = wmma16(cat8(alo, ahi), b, c);
  }

  int n = n0 + ln;
  float bn = bias[n];
#pragma unroll
  for (int r = 0; r < 8; r++) {
    int m = row0 + r + 8 * hs;                 // D-frag: (M = r + 8*hi, N = ln)
    float v = c[r] + bn;
    if (mode == 0) {
      outH[(size_t)m * DIM + n] = (_Float16)v;
    } else if (mode == 1) {
      int bb = m >> 10, s = m & (SEQ - 1);
      outH[((size_t)bb * DIM + n) * SEQ + s] = (_Float16)v;
    } else {
      outF[(size_t)m * DIM + n] = v;
    }
  }
}

// Q, K, V projections fused into one launch: grid = 3 * 256 blocks.
__global__ __launch_bounds__(256) void proj_qkv(
    const _Float16* __restrict__ xh,
    const _Float16* __restrict__ Wqh, const _Float16* __restrict__ Wkh,
    const _Float16* __restrict__ Wvh,
    const float* __restrict__ bq, const float* __restrict__ bk,
    const float* __restrict__ bv,
    _Float16* __restrict__ Qh, _Float16* __restrict__ Kh,
    _Float16* __restrict__ Vt) {
  int sel = blockIdx.x >> 8;      // 0:Q 1:K 2:V
  int blk = blockIdx.x & 255;
  if (sel == 0)      proj_tile(xh, Wqh, bq, Qh, nullptr, 0, blk);
  else if (sel == 1) proj_tile(xh, Wkh, bk, Kh, nullptr, 0, blk);
  else               proj_tile(xh, Wvh, bv, Vt, nullptr, 1, blk);
}

__global__ __launch_bounds__(256) void proj_out(
    const _Float16* __restrict__ attH, const _Float16* __restrict__ Woh,
    const float* __restrict__ bo, float* __restrict__ out) {
  proj_tile(attH, Woh, bo, nullptr, out, 2, blockIdx.x);
}

// ----------------------- fused geodesic flash attention --------------------
// grid = BATCH * (SEQ/16) = 128 blocks, 64 threads (2 waves) -> same per-wave
// work as a 64-row tile but 4x the WGP spread. Loop over 32-key tiles:
//   S = Q Kt^T (WMMA); p = exp(-sqrt(|q|^2+|k|^2-2qk)/16); P -> LDS;
//   rowsums via half-wave shfl reductions (deterministic 2-term combine);
//   acc += P @ V (WMMA, A-frags from LDS, B-frags from D-major V).
// Scores <= 0 so exp never overflows: no online-max rescale needed.
#define PSTRIDE 40   // padded LDS row stride (f16) to spread banks
__global__ __launch_bounds__(64) void attn_kernel(
    const _Float16* __restrict__ Qh, const _Float16* __restrict__ Kh,
    const _Float16* __restrict__ Vt, const float* __restrict__ qn,
    const float* __restrict__ kn, _Float16* __restrict__ attH) {
  __shared__ __align__(16) _Float16 Plds[16 * PSTRIDE];
  __shared__ float partLds[16][2];
  __shared__ float rowsum_s[16];

  int b  = blockIdx.x >> 6;          // SEQ/16 = 64 query tiles per batch
  int q0 = (blockIdx.x & 63) * 16;   // query tile base within batch
  int tid = threadIdx.x;
  int ni  = tid >> 5;                // QK: key sub-tile; PV: half of D
  int lane = tid & 31;
  int hs = lane >> 4, ln = lane & 15;

  const _Float16* Qb = Qh + (size_t)b * SEQ * DIM;
  const _Float16* Kb = Kh + (size_t)b * SEQ * DIM;
  const _Float16* Vb = Vt + (size_t)b * DIM * SEQ;
  const float* qnb = qn + b * SEQ;
  const float* knb = kn + b * SEQ;

  float qnr[8];
#pragma unroll
  for (int r = 0; r < 8; r++) qnr[r] = qnb[q0 + r + 8 * hs];

  const _Float16* Arow = Qb + (size_t)(q0 + ln) * DIM + hs * 16;

  v8f zero8 = {};
  v8f acc[8];
#pragma unroll
  for (int t = 0; t < 8; t++) acc[t] = zero8;

  float rsum = 0.f;  // valid for tid < 16 (one thread per query row)

  for (int kt0 = 0; kt0 < SEQ; kt0 += 32) {
    // ---- score tile: 16 queries x 16 keys per wave, K = 256 ----
    const _Float16* Brow = Kb + (size_t)(kt0 + ni * 16 + ln) * DIM + hs * 16;
    if (kt0 + 32 < SEQ) {  // global_prefetch_b8 of next K and V tiles
      __builtin_prefetch(Kb + (size_t)(kt0 + 32 + ni * 16 + ln) * DIM + hs * 16);
      __builtin_prefetch(Vb + (size_t)(ni * 128 + ln) * SEQ + kt0 + 32 + hs * 16);
    }
    v8f c = {};
#pragma unroll
    for (int k0 = 0; k0 < DIM; k0 += 32) {
      v8h alo = *(const v8h*)(Arow + k0);
      v8h ahi = *(const v8h*)(Arow + k0 + 8);
      v16h bf = *(const v16h*)(Brow + k0);
      c = wmma16(cat8(alo, ahi), bf, c);
    }
    float knv = knb[kt0 + ni * 16 + ln];

    __syncthreads();  // previous P tile fully consumed
#pragma unroll
    for (int r = 0; r < 8; r++) {
      float d2 = qnr[r] + knv - 2.f * c[r];
      d2 = d2 > 0.f ? d2 : 0.f;
      float pf = __expf(-__builtin_sqrtf(d2) * 0.0625f);  // -dist/sqrt(256)
      _Float16 ph = (_Float16)pf;
      Plds[(r + 8 * hs) * PSTRIDE + ni * 16 + ln] = ph;
      // half-wave (16-lane) reduction over keys of this sub-tile
      float s = (float)ph;
#pragma unroll
      for (int off = 8; off; off >>= 1) s += __shfl_xor(s, off, 16);
      if (ln == 0) partLds[r + 8 * hs][ni] = s;
    }
    __syncthreads();  // P tile + partial sums ready

    if (tid < 16) rsum += partLds[tid][0] + partLds[tid][1];  // fixed order

    // ---- PV: acc(16 x 128 per wave) += P(16x32) @ V(32x128) ----
    const _Float16* pa = &Plds[ln * PSTRIDE + hs * 16];
    v8h plo = *(const v8h*)(pa);
    v8h phi = *(const v8h*)(pa + 8);
    v16h pfrag = cat8(plo, phi);
#pragma unroll
    for (int t = 0; t < 8; t++) {
      int n = ni * 128 + t * 16 + ln;                 // embedding column
      v16h vf = *(const v16h*)(Vb + (size_t)n * SEQ + kt0 + hs * 16);
      acc[t] = wmma16(pfrag, vf, acc[t]);
    }
  }

  if (tid < 16) rowsum_s[tid] = rsum;
  __syncthreads();

  float rinv[8];
#pragma unroll
  for (int r = 0; r < 8; r++) rinv[r] = 1.f / rowsum_s[r + 8 * hs];

  _Float16* outb = attH + ((size_t)b * SEQ + q0) * DIM;
#pragma unroll
  for (int t = 0; t < 8; t++) {
    int n = ni * 128 + t * 16 + ln;
#pragma unroll
    for (int r = 0; r < 8; r++)
      outb[(size_t)(r + 8 * hs) * DIM + n] = (_Float16)(acc[t][r] * rinv[r]);
  }
}

// ---------------------------------------------------------------------------
extern "C" void kernel_launch(void* const* d_in, const int* in_sizes, int n_in,
                              void* d_out, int out_size, void* d_ws, size_t ws_size,
                              hipStream_t stream) {
  (void)in_sizes; (void)n_in; (void)out_size; (void)ws_size;
  const float* x  = (const float*)d_in[0];
  const float* Wq = (const float*)d_in[1];
  const float* bq = (const float*)d_in[2];
  const float* Wk = (const float*)d_in[3];
  const float* bk = (const float*)d_in[4];
  const float* Wv = (const float*)d_in[5];
  const float* bv = (const float*)d_in[6];
  const float* Wo = (const float*)d_in[7];
  const float* bo = (const float*)d_in[8];

  char* ws = (char*)d_ws;
  size_t off = 0;
  auto take = [&](size_t bytes) -> char* {
    char* p = ws + off;
    off += (bytes + 255) & ~(size_t)255;
    return p;
  };
  _Float16* xh   = (_Float16*)take((size_t)NX * 2);
  _Float16* Wqh  = (_Float16*)take((size_t)NW * 2);
  _Float16* Wkh  = (_Float16*)take((size_t)NW * 2);
  _Float16* Wvh  = (_Float16*)take((size_t)NW * 2);
  _Float16* Woh  = (_Float16*)take((size_t)NW * 2);
  _Float16* Qhp  = (_Float16*)take((size_t)NX * 2);
  _Float16* Khp  = (_Float16*)take((size_t)NX * 2);
  _Float16* Vtp  = (_Float16*)take((size_t)NX * 2);  // D-major (transposed)
  _Float16* attH = (_Float16*)take((size_t)NX * 2);
  float*    qn   = (float*)take((size_t)MROWS * 4);
  float*    kn   = (float*)take((size_t)MROWS * 4);

  // 1) fp32 -> f16 operands (x + all weights), one launch
  cvt_all<<<(NX + 4 * NW) / 256, 256, 0, stream>>>(
      x, Wq, Wk, Wv, Wo, xh, Wqh, Wkh, Wvh, Woh);

  // 2) Q/K/V projections (WMMA GEMMs) fused into one launch
  proj_qkv<<<3 * 256, 256, 0, stream>>>(
      xh, Wqh, Wkh, Wvh, bq, bk, bv, Qhp, Khp, Vtp);

  // 3) row squared norms of Q and K (||q-k||^2 expansion), one launch
  norms_kernel<<<2 * MROWS / 8, 256, 0, stream>>>(Qhp, Khp, qn, kn);

  // 4) fused distance-softmax-PV attention (128 blocks x 2 waves)
  attn_kernel<<<BATCH * (SEQ / 16), 64, 0, stream>>>(
      Qhp, Khp, Vtp, qn, kn, attH);

  // 5) output projection to f32
  proj_out<<<256, 256, 0, stream>>>(attH, Woh, bo, (float*)d_out);
}
